// SelfAttentionWithGate_31885837206278
// MI455X (gfx1250) — matrix operand order repaired and checked by reading, not compiled, not run on hardware
//
#include <hip/hip_runtime.h>
#include <hip/hip_bf16.h>

// ---------------------------------------------------------------------------
// SelfAttentionWithGate for MI455X (gfx1250, wave32, WMMA bf16 16x16x32)
//   B=8, S=2048, D_MODEL=1024, TDS=64, DIN=1088, DH=512
// ---------------------------------------------------------------------------

#define BATCH   8
#define SEQ     2048
#define DMODEL  1024
#define TDS     64
#define DIN     1088
#define DH      512

typedef __attribute__((ext_vector_type(16))) __bf16 v16bf;
typedef __attribute__((ext_vector_type(8)))  __bf16 v8bf;
typedef __attribute__((ext_vector_type(8)))  float  v8f;

__device__ __forceinline__ unsigned short f2bf(float f) {
    union { __bf16 h; unsigned short u; } cv;
    cv.h = (__bf16)f;            // hardware f32->bf16 convert
    return cv.u;
}

__device__ __forceinline__ v16bf combine16(v8bf lo, v8bf hi) {
    return __builtin_shufflevector(lo, hi, 0,1,2,3,4,5,6,7,8,9,10,11,12,13,14,15);
}

__device__ __forceinline__ v8f vzero8() {
    v8f z;
#pragma unroll
    for (int i = 0; i < 8; ++i) z[i] = 0.0f;
    return z;
}

__device__ __forceinline__ v8f wmma_bf16(v16bf a, v16bf b, v8f c) {
    return __builtin_amdgcn_wmma_f32_16x16x32_bf16(false, a, false, b, (short)0, c, false, false);
}

// ---------------------------------------------------------------------------
// Kernel 1: QKV projection.  x = concat(tf, env) [16384 x 1088] (f32 -> bf16)
// out[sel] = bf16( x @ W[sel] + bias[sel] )   for sel in {Q,K,V}
// grid = (M/128, 512/64, 3), block = 256 (8 waves, each wave 16 rows x 64 cols)
// ---------------------------------------------------------------------------
__global__ __launch_bounds__(256)
void qkv_proj_kernel(const float* __restrict__ tf, const float* __restrict__ env,
                     const float* __restrict__ Wq, const float* __restrict__ bq,
                     const float* __restrict__ Wk, const float* __restrict__ bk,
                     const float* __restrict__ Wv, const float* __restrict__ bv,
                     unsigned short* __restrict__ Qb,
                     unsigned short* __restrict__ Kb,
                     unsigned short* __restrict__ Vb)
{
    const float* W;  const float* bias;  unsigned short* outp;
    if      (blockIdx.z == 0) { W = Wq; bias = bq; outp = Qb; }
    else if (blockIdx.z == 1) { W = Wk; bias = bk; outp = Kb; }
    else                      { W = Wv; bias = bv; outp = Vb; }

    __shared__ alignas(32) unsigned short As[128 * 32];  // [row][k]
    __shared__ alignas(32) unsigned short Bs[64 * 32];   // [n][k]

    const int tid  = threadIdx.x;
    const int lane = tid & 31;
    const int wv   = tid >> 5;
    const int lrow = lane & 15;
    const int hi16 = (lane >> 4) & 1;
    const int m0   = blockIdx.x * 128;
    const int n0   = blockIdx.y * 64;

    v8f acc[4] = { vzero8(), vzero8(), vzero8(), vzero8() };

    for (int kt = 0; kt < DIN / 32; ++kt) {
        const int k0 = kt * 32;
        // ---- stage A tile (concat + f32->bf16) ----
        {
            const int  row  = tid >> 1;          // 0..127
            const int  kk0  = (tid & 1) * 16;
            const long gbase = (long)(m0 + row) * DMODEL;
            const int  bidx  = (m0 + row) >> 11; // /SEQ
#pragma unroll
            for (int i = 0; i < 16; ++i) {
                const int k = k0 + kk0 + i;
                const float v = (k < DMODEL) ? tf[gbase + k]
                                             : env[bidx * TDS + (k - DMODEL)];
                As[row * 32 + kk0 + i] = f2bf(v);
            }
        }
        // ---- stage B tile ([n][k], f32->bf16) ----
        {
            const int n   = tid >> 2;            // 0..63
            const int kk0 = (tid & 3) * 8;
#pragma unroll
            for (int i = 0; i < 8; ++i) {
                const int k = k0 + kk0 + i;
                Bs[n * 32 + kk0 + i] = f2bf(W[(long)k * DH + (n0 + n)]);
            }
        }
        __syncthreads();

        // ---- A fragment: row = wv*16 + lrow; K lo/hi per lane half ----
        const unsigned short* arow = As + (wv * 16 + lrow) * 32 + (hi16 ? 16 : 0);
        const v8bf alo = *(const v8bf*)(arow);
        const v8bf ahi = *(const v8bf*)(arow + 8);
        const v16bf a  = combine16(alo, ahi);

#pragma unroll
        for (int t = 0; t < 4; ++t) {
            const v16bf bf = *(const v16bf*)(Bs + (t * 16 + lrow) * 32 + (hi16 ? 16 : 0));
            acc[t] = wmma_bf16(a, bf, acc[t]);
        }
        __syncthreads();
    }

    // ---- epilogue: + bias, f32->bf16 store ----
#pragma unroll
    for (int t = 0; t < 4; ++t) {
        const int n = n0 + t * 16 + lrow;
        const float bval = bias[n];
#pragma unroll
        for (int r = 0; r < 8; ++r) {
            const int row = m0 + wv * 16 + r + (hi16 ? 8 : 0);
            outp[(long)row * DH + n] = f2bf(acc[t][r] + bval);
        }
    }
}

// ---------------------------------------------------------------------------
// Kernel 2: gate = sigmoid(x @ Wg + bg).  One wave per row.
// ---------------------------------------------------------------------------
__global__ __launch_bounds__(256)
void gate_kernel(const float* __restrict__ tf, const float* __restrict__ env,
                 const float* __restrict__ Wg, const float* __restrict__ bg,
                 float* __restrict__ gateb)
{
    const int  wv   = threadIdx.x >> 5;
    const int  lane = threadIdx.x & 31;
    const long row  = (long)blockIdx.x * 8 + wv;
    const int  bidx = (int)(row >> 11);

    const float* trow = tf + row * DMODEL;
    float sum = 0.0f;
    for (int k = lane; k < DMODEL; k += 32) sum += trow[k] * Wg[k];
    for (int k = DMODEL + lane; k < DIN; k += 32)
        sum += env[bidx * TDS + (k - DMODEL)] * Wg[k];
#pragma unroll
    for (int mask = 16; mask; mask >>= 1) sum += __shfl_xor(sum, mask, 32);
    if (lane == 0)
        gateb[row] = 1.0f / (1.0f + __expf(-(sum + bg[0])));
}

// ---------------------------------------------------------------------------
// Kernel 3: flash attention.
// Block = 1 batch x 32 queries. 8 waves = 2 query-groups x 4 d-slices (128).
// Key blocks of 64; online softmax in WMMA C layout.
// K tile staged via GLOBAL_LOAD_ASYNC_TO_LDS (ASYNCcnt); V transposed via LDS.
// LDS: Ks[64][512] bf16 | Vt[512][64] bf16 | sS[2][16][64] f32 | pS[2][16][64] bf16
// ---------------------------------------------------------------------------
#define KBK 64

__global__ __launch_bounds__(256)
void attn_kernel(const unsigned short* __restrict__ Qb,
                 const unsigned short* __restrict__ Kb,
                 const unsigned short* __restrict__ Vb,
                 const float* __restrict__ gateb,
                 float* __restrict__ out)
{
    extern __shared__ char smem[];
    unsigned short* Ks = (unsigned short*)(smem);            // 64*512*2  = 65536
    unsigned short* Vt = (unsigned short*)(smem + 65536);    // 512*64*2  = 65536
    float*          sS = (float*)(smem + 131072);            // 2*16*64*4 = 8192
    unsigned short* pS = (unsigned short*)(smem + 139264);   // 2*16*64*2 = 4096

    const int tid  = threadIdx.x;
    const int lane = tid & 31;
    const int wv   = tid >> 5;
    const int g    = wv >> 2;        // query group 0..1
    const int h    = wv & 3;         // d-slice 0..3
    const int dh0  = h * 128;
    const int lrow = lane & 15;
    const int hi16 = (lane >> 4) & 1;

    const int  b    = blockIdx.x >> 6;
    const int  q0   = (blockIdx.x & 63) * 32;
    const long rowg = (long)b * SEQ + q0 + g * 16;

    // ---- preload Q fragments for this wave's 16 queries x 128 d-slice ----
    v16bf aq[4];
    {
        const unsigned short* qrow = Qb + (rowg + lrow) * DH;
#pragma unroll
        for (int c = 0; c < 4; ++c) {
            const int dbase = dh0 + c * 32;
            const v8bf lo = *(const v8bf*)(qrow + dbase + (hi16 ? 16 : 0));
            const v8bf hi = *(const v8bf*)(qrow + dbase + 8 + (hi16 ? 16 : 0));
            aq[c] = combine16(lo, hi);
        }
    }

    float gate_r[8], m_r[8], l_r[8];
#pragma unroll
    for (int r = 0; r < 8; ++r) {
        gate_r[r] = gateb[rowg + r + hi16 * 8] * 0.03125f;  // gate / sqrt(1024)
        m_r[r] = -1e30f;
        l_r[r] = 0.0f;
    }
    v8f oacc[8];
#pragma unroll
    for (int t = 0; t < 8; ++t) oacc[t] = vzero8();

    const unsigned short* Kbase = Kb + (long)b * SEQ * DH;
    const unsigned short* Vbase = Vb + (long)b * SEQ * DH;
    const unsigned ks_lds = (unsigned)(unsigned long)(void*)Ks;  // LDS byte offset

    for (int kb = 0; kb < SEQ / KBK; ++kb) {
        // ---- zero score accumulator ----
#pragma unroll
        for (int i = 0; i < 8; ++i) sS[tid * 8 + i] = 0.0f;

        // ---- stage K block: async DMA global -> LDS (bypasses VGPRs) ----
        {
            const unsigned short* src = Kbase + (long)kb * KBK * DH;
#pragma unroll
            for (int j = 0; j < 16; ++j) {
                const int cid = tid + 256 * j;                    // 16B chunk id
                unsigned long gaddr = (unsigned long)(src + cid * 8);
                unsigned      laddr = ks_lds + cid * 16;
                asm volatile("global_load_async_to_lds_b128 %0, %1, off"
                             :: "v"(laddr), "v"(gaddr) : "memory");
            }
        }
        // ---- stage V block transposed: Vt[d][key], packed pair stores ----
        {
            const uint4* src = (const uint4*)(Vbase + (long)kb * KBK * DH);
            unsigned*    VtW = (unsigned*)Vt;
#pragma unroll
            for (int j = 0; j < 8; ++j) {
                const int pid = tid + 256 * j;     // 0..2047
                const int kk2 = pid >> 6;          // key pair 0..31
                const int dq  = pid & 63;          // d octet
                uint4 c0 = src[(kk2 * 2    ) * 64 + dq];
                uint4 c1 = src[(kk2 * 2 + 1) * 64 + dq];
                const unsigned short* e0 = (const unsigned short*)&c0;
                const unsigned short* e1 = (const unsigned short*)&c1;
#pragma unroll
                for (int q = 0; q < 8; ++q)
                    VtW[(dq * 8 + q) * 32 + kk2] =
                        (unsigned)e0[q] | ((unsigned)e1[q] << 16);
            }
        }
        // ---- prefetch next key block into L2 while we compute ----
        if (kb + 1 < SEQ / KBK) {
            const char* nk = (const char*)(Kbase + (long)(kb + 1) * KBK * DH);
            const char* nv = (const char*)(Vbase + (long)(kb + 1) * KBK * DH);
            __builtin_prefetch(nk + tid * 256, 0, 0);
            __builtin_prefetch(nv + tid * 256, 0, 0);
        }
        asm volatile("s_wait_asynccnt 0x0" ::: "memory");
        __syncthreads();

        // ---- partial scores over this wave's 128-d slice ----
        v8f acc4[4] = { vzero8(), vzero8(), vzero8(), vzero8() };
#pragma unroll
        for (int c = 0; c < 4; ++c) {
#pragma unroll
            for (int t = 0; t < 4; ++t) {
                const v16bf bf = *(const v16bf*)(Ks + (t * 16 + lrow) * DH +
                                                 dh0 + c * 32 + (hi16 ? 16 : 0));
                acc4[t] = wmma_bf16(aq[c], bf, acc4[t]);
            }
        }
        // reduce partials across the 4 d-slice waves
#pragma unroll
        for (int t = 0; t < 4; ++t)
#pragma unroll
            for (int r = 0; r < 8; ++r)
                atomicAdd(&sS[(g * 16 + r + hi16 * 8) * KBK + t * 16 + lrow], acc4[t][r]);
        __syncthreads();

        // ---- online softmax (redundant across the 4 h-waves, deterministic) ----
        float s[4][8];
#pragma unroll
        for (int t = 0; t < 4; ++t)
#pragma unroll
            for (int r = 0; r < 8; ++r)
                s[t][r] = sS[(g * 16 + r + hi16 * 8) * KBK + t * 16 + lrow] * gate_r[r];

        float tm[8];
#pragma unroll
        for (int r = 0; r < 8; ++r)
            tm[r] = fmaxf(fmaxf(s[0][r], s[1][r]), fmaxf(s[2][r], s[3][r]));
#pragma unroll
        for (int mask = 1; mask <= 8; mask <<= 1)
#pragma unroll
            for (int r = 0; r < 8; ++r)
                tm[r] = fmaxf(tm[r], __shfl_xor(tm[r], mask, 32));

        float corr[8];
#pragma unroll
        for (int r = 0; r < 8; ++r) {
            const float mn = fmaxf(m_r[r], tm[r]);
            corr[r] = __expf(m_r[r] - mn);
            m_r[r]  = mn;
        }
        float p[4][8], rs[8];
#pragma unroll
        for (int r = 0; r < 8; ++r) rs[r] = 0.0f;
#pragma unroll
        for (int t = 0; t < 4; ++t)
#pragma unroll
            for (int r = 0; r < 8; ++r) {
                p[t][r] = __expf(s[t][r] - m_r[r]);
                rs[r] += p[t][r];
            }
#pragma unroll
        for (int mask = 1; mask <= 8; mask <<= 1)
#pragma unroll
            for (int r = 0; r < 8; ++r)
                rs[r] += __shfl_xor(rs[r], mask, 32);
#pragma unroll
        for (int r = 0; r < 8; ++r) l_r[r] = l_r[r] * corr[r] + rs[r];
#pragma unroll
        for (int t = 0; t < 8; ++t)
#pragma unroll
            for (int r = 0; r < 8; ++r) oacc[t][r] *= corr[r];

        if (h == 0) {
#pragma unroll
            for (int t = 0; t < 4; ++t)
#pragma unroll
                for (int r = 0; r < 8; ++r)
                    pS[(g * 16 + r + hi16 * 8) * KBK + t * 16 + lrow] = f2bf(p[t][r]);
        }
        __syncthreads();

        // ---- out += p @ V  (this wave's 128-d slice) ----
#pragma unroll
        for (int c2 = 0; c2 < 2; ++c2) {
            const unsigned short* prow = pS + (g * 16 + lrow) * KBK + c2 * 32;
            const v16bf ap = combine16(*(const v8bf*)(prow + (hi16 ? 16 : 0)),
                                       *(const v8bf*)(prow + 8 + (hi16 ? 16 : 0)));
#pragma unroll
            for (int nt = 0; nt < 8; ++nt) {
                const v16bf bf = *(const v16bf*)(Vt + (dh0 + nt * 16 + lrow) * KBK +
                                                 c2 * 32 + (hi16 ? 16 : 0));
                oacc[nt] = wmma_bf16(ap, bf, oacc[nt]);
            }
        }
        __syncthreads();  // protect Ks/Vt/pS/sS before next iteration restages
    }

    // ---- epilogue: normalize and store f32 ----
    float invl[8];
#pragma unroll
    for (int r = 0; r < 8; ++r) invl[r] = 1.0f / l_r[r];
#pragma unroll
    for (int nt = 0; nt < 8; ++nt)
#pragma unroll
        for (int r = 0; r < 8; ++r)
            out[(rowg + r + hi16 * 8) * DH + dh0 + nt * 16 + lrow] = oacc[nt][r] * invl[r];
}

// ---------------------------------------------------------------------------
extern "C" void kernel_launch(void* const* d_in, const int* in_sizes, int n_in,
                              void* d_out, int out_size, void* d_ws, size_t ws_size,
                              hipStream_t stream) {
    const float* tf  = (const float*)d_in[0];
    const float* env = (const float*)d_in[1];
    const float* Wq  = (const float*)d_in[2];
    const float* bq  = (const float*)d_in[3];
    const float* Wk  = (const float*)d_in[4];
    const float* bk  = (const float*)d_in[5];
    const float* Wv  = (const float*)d_in[6];
    const float* bv  = (const float*)d_in[7];
    const float* Wg  = (const float*)d_in[8];
    const float* bg  = (const float*)d_in[9];
    float* out = (float*)d_out;

    char* ws = (char*)d_ws;
    unsigned short* Qb    = (unsigned short*)(ws);
    unsigned short* Kb    = (unsigned short*)(ws + (size_t)16 * 1024 * 1024);
    unsigned short* Vb    = (unsigned short*)(ws + (size_t)32 * 1024 * 1024);
    float*          gateb = (float*)        (ws + (size_t)48 * 1024 * 1024);

    dim3 gproj(128, 8, 3);  // (16384/128) x (512/64) x {Q,K,V}
    qkv_proj_kernel<<<gproj, 256, 0, stream>>>(tf, env, Wq, bq, Wk, bk, Wv, bv,
                                               Qb, Kb, Vb);
    gate_kernel<<<BATCH * SEQ / 8, 256, 0, stream>>>(tf, env, Wg, bg, gateb);

    const size_t smem = 65536 + 65536 + 8192 + 4096;  // 143360 B
    hipFuncSetAttribute(reinterpret_cast<const void*>(attn_kernel),
                        hipFuncAttributeMaxDynamicSharedMemorySize, (int)smem);
    attn_kernel<<<BATCH * (SEQ / 32), 256, smem, stream>>>(Qb, Kb, Vb, gateb, out);
}
